// FractalOpponent_32538672234982
// MI455X (gfx1250) — matrix operand order repaired and checked by reading, not compiled, not run on hardware
//
// MI455X (gfx1250, CDNA5) implementation of the recursive fractal LSTM cell.
//
// Strategy: batch-1 GEMV workload -> bandwidth bound.
//  * Convert all hot weights fp32 -> f16 ONCE per launch, pre-swizzled into
//    the WMMA B-fragment lane layout. f16 working set = 144 MB < 192 MB L2,
//    so all 5 recursion passes are served from L2 instead of HBM.
//  * GEMV via v_wmma_f32_16x16x32_f16: A = activation vector replicated over
//    all 16 M rows; each WMMA consumes a 1KB weight tile with two b128 loads.
//    Inner loop unrolled x4 (16 loads batched ahead of 4 chained WMMAs) to
//    tolerate L2 latency at low occupancy; wave-uniform control is forced
//    into SGPRs via readfirstlane for scalar loop control.
//  * K split across waves, fp32 global atomic-add reduction.
//  * Data-dependent recursion handled by a fully unrolled worst-case launch
//    schedule; every kernel is predicated on a device-side control word
//    (steps per frame), computed by tiny on-device router/argmax kernels.
//    Deterministic, graph-capture safe (no host syncs, no allocs).
//
// Workspace requirement: ~146 MB (f16 weights + small vector arena).

#include <hip/hip_runtime.h>
#include <hip/hip_bf16.h>
#include <math.h>

#define HD   2048
#define HD4  8192
#define KK2  4096
#define BIG  (1<<30)

typedef __attribute__((ext_vector_type(16))) _Float16 v16h;
typedef __attribute__((ext_vector_type(8)))  float    v8f;

union FragH { uint4 q[2]; v16h v; };

#define PRED const int* __restrict__ ctl, int ci, int lo, int hi
#define PCHECK do { if (ci >= 0) { int s_ = ctl[ci]; if (s_ < lo || s_ > hi) return; } } while (0)

__device__ __forceinline__ unsigned pack2h(float a, float b) {
  _Float16 ha = (_Float16)a, hb = (_Float16)b;
  unsigned short ua, ub;
  __builtin_memcpy(&ua, &ha, 2);
  __builtin_memcpy(&ub, &hb, 2);
  return (unsigned)ua | ((unsigned)ub << 16);
}
__device__ __forceinline__ float sigf(float x) { return 1.0f / (1.0f + expf(-x)); }

// ---------------------------------------------------------------------------
// Weight conversion fp32 row-major (K x N) -> f16 swizzled 32x16 tiles.
// Tile (kt,nt): 32 lanes x 8 dwords (1KB). Lane l, dword v holds the packed
// pair { W[k0+kk][n0+c], W[k0+kk+1][n0+c] }, c = l&15, kk = 2v + (l>=16 ? 16:0)
// -- contiguous k order (k = 16h + s), the same order k_gemv uses for A.
// ---------------------------------------------------------------------------
__global__ void k_convert(const float* __restrict__ W, unsigned* __restrict__ out,
                          int K, int N) {
  int tid = blockIdx.x * 256 + threadIdx.x;
  int total = (K >> 1) * N;             // output dwords
  if (tid >= total) return;
  int nTiles = N >> 4;
  int tile = tid >> 8, r = tid & 255, lane = r >> 3, v = r & 7;
  int kt = tile / nTiles, nt = tile % nTiles;
  int k = (kt << 5) + (v << 1) + ((lane >> 4) << 4);
  int n = (nt << 4) + (lane & 15);
  out[tid] = pack2h(W[k * N + n], W[(k + 1) * N + n]);
}

// ---------------------------------------------------------------------------
// y[n] += v . W[:,n]  (W pre-swizzled f16, v f16).  One wave per 16-col tile
// per K-chunk; K-chunks reduced with global fp32 atomics (y zeroed first).
// A-fragment: vector replicated across all 16 M rows => every D row == y,
// lanes 0..15 read acc[0] = y[n0+lane].  ktPer must be a multiple of 4.
// ---------------------------------------------------------------------------
__global__ void k_gemv(PRED,
                       const unsigned* __restrict__ Wt,
                       const _Float16* __restrict__ vec,
                       float* __restrict__ y,
                       int K, int N, int kChunks) {
  PCHECK;
  int lane = threadIdx.x & 31;
  // wave-uniform control -> SGPRs for scalar loop control / saddr loads
  int wave = __builtin_amdgcn_readfirstlane(
      (int)(blockIdx.x * (blockDim.x >> 5) + (threadIdx.x >> 5)));
  int nTiles = __builtin_amdgcn_readfirstlane(N >> 4);
  if (wave >= nTiles * kChunks) return;
  int nt  = __builtin_amdgcn_readfirstlane(wave % nTiles);
  int kc  = __builtin_amdgcn_readfirstlane(wave / nTiles);
  int ktTot = K >> 5;
  int ktPer = __builtin_amdgcn_readfirstlane(ktTot / kChunks);
  int kt0   = __builtin_amdgcn_readfirstlane(kc * ktPer);
  int half  = (lane >= 16) ? 16 : 0;

  v8f acc = {};
  const uint4* tp = (const uint4*)Wt + (size_t)(kt0 * nTiles + nt) * 64 + lane * 2;
  const size_t tileStride = (size_t)nTiles * 64;   // uint4 per K-tile step

  for (int kt = kt0; kt < kt0 + ktPer; kt += 4) {
    FragH A[4], B[4];
#pragma unroll
    for (int u = 0; u < 4; ++u) {
      const uint4* tpu = tp + (size_t)u * tileStride;
      B[u].q[0] = tpu[0];
      B[u].q[1] = tpu[1];
      int kb = ((kt + u) << 5) + half;
      A[u].q[0] = *(const uint4*)(vec + kb);       // v[kb .. kb+7]
      A[u].q[1] = *(const uint4*)(vec + kb + 8);   // v[kb+8 .. kb+15]
    }
#pragma unroll
    for (int u = 0; u < 4; ++u)
      acc = __builtin_amdgcn_wmma_f32_16x16x32_f16(false, A[u].v, false, B[u].v,
                                                   (short)0, acc, false, false);
    tp += 4 * tileStride;
    __builtin_prefetch((const void*)tp, 0, 1);     // global_prefetch_b8
  }
  if (lane < 16) atomicAdd(&y[(nt << 4) + lane], acc[0]);
}

// ------------------------------- small kernels -----------------------------
__global__ void k_ctlinit(int* ctl) { if (threadIdx.x < 16) ctl[threadIdx.x] = 0; }

__global__ void k_zero(PRED, float* p, int n) {
  PCHECK;
  int i = blockIdx.x * 256 + threadIdx.x;
  if (i < n) p[i] = 0.0f;
}
__global__ void k_cvt(PRED, const float* __restrict__ in, _Float16* __restrict__ out, int n) {
  PCHECK;
  int i = blockIdx.x * 256 + threadIdx.x;
  if (i < n) out[i] = (_Float16)in[i];
}
__global__ void k_copy(PRED, const float* __restrict__ in, float* __restrict__ out,
                       _Float16* outh, int n) {
  PCHECK;
  int i = blockIdx.x * 256 + threadIdx.x;
  if (i < n) {
    float v = in[i];
    out[i] = v;
    if (outh) outh[i] = (_Float16)v;
  }
}

// gates (already x@Wih + attn@Whh) + b_lstm -> LSTM pointwise; so = hs = h2.
__global__ void k_lstm(PRED, const float* __restrict__ gates, const float* __restrict__ b,
                       const float* __restrict__ cin,
                       float* __restrict__ h2, _Float16* __restrict__ h2h,
                       float* __restrict__ c2) {
  PCHECK;
  int i = blockIdx.x * 256 + threadIdx.x;
  if (i >= HD) return;
  float gi = gates[i]          + b[i];
  float gf = gates[HD + i]     + b[HD + i];
  float gg = gates[2 * HD + i] + b[2 * HD + i];
  float go = gates[3 * HD + i] + b[3 * HD + i];
  float c = sigf(gf) * cin[i] + sigf(gi) * tanhf(gg);
  float h = sigf(go) * tanhf(c);
  c2[i] = c;
  h2[i] = h;
  h2h[i] = (_Float16)h;
}

// p = softmax(hs @ router_w + router_b); single block, LDS tree reduction.
__global__ void k_router(PRED, const float* __restrict__ hs,
                         const float* __restrict__ rw, const float* __restrict__ rb,
                         float* __restrict__ p) {
  PCHECK;
  __shared__ float red[4][256];
  int t = threadIdx.x;
  float acc[4] = {0.f, 0.f, 0.f, 0.f};
  for (int k = t; k < HD; k += 256) {
    float hv = hs[k];
    acc[0] += hv * rw[k * 4 + 0];
    acc[1] += hv * rw[k * 4 + 1];
    acc[2] += hv * rw[k * 4 + 2];
    acc[3] += hv * rw[k * 4 + 3];
  }
  for (int e = 0; e < 4; ++e) red[e][t] = acc[e];
  __syncthreads();
  for (int s = 128; s > 0; s >>= 1) {
    if (t < s)
      for (int e = 0; e < 4; ++e) red[e][t] += red[e][t + s];
    __syncthreads();
  }
  if (t == 0) {
    float l[4], m = -1e30f, sum = 0.f;
    for (int e = 0; e < 4; ++e) { l[e] = red[e][0] + rb[e]; m = fmaxf(m, l[e]); }
    for (int e = 0; e < 4; ++e) { l[e] = expf(l[e] - m); sum += l[e]; }
    for (int e = 0; e < 4; ++e) p[e] = l[e] / sum;
  }
}

// hp = sum_e p[e] * (eo[e] + expert_b[e])
__global__ void k_expcomb(PRED, const float* __restrict__ eo, const float* __restrict__ p,
                          const float* __restrict__ eb,
                          float* __restrict__ hp, _Float16* __restrict__ hph) {
  PCHECK;
  int i = blockIdx.x * 256 + threadIdx.x;
  if (i >= HD) return;
  float s = 0.f;
  for (int e = 0; e < 4; ++e) s += p[e] * (eo[e * HD + i] + eb[e * HD + i]);
  hp[i] = s;
  hph[i] = (_Float16)s;
}

// ctl[slot] = min(argmax(hp @ dr_w + dr_b), maxSteps); single block.
__global__ void k_depth(PRED, const float* __restrict__ hp,
                        const float* __restrict__ drw, const float* __restrict__ drb,
                        int maxSteps, int slot, int* __restrict__ ctlw) {
  PCHECK;
  __shared__ float red[3][256];
  int t = threadIdx.x;
  float acc[3] = {0.f, 0.f, 0.f};
  for (int k = t; k < HD; k += 256) {
    float hv = hp[k];
    acc[0] += hv * drw[k * 3 + 0];
    acc[1] += hv * drw[k * 3 + 1];
    acc[2] += hv * drw[k * 3 + 2];
  }
  for (int j = 0; j < 3; ++j) red[j][t] = acc[j];
  __syncthreads();
  for (int s = 128; s > 0; s >>= 1) {
    if (t < s)
      for (int j = 0; j < 3; ++j) red[j][t] += red[j][t + s];
    __syncthreads();
  }
  if (t == 0) {
    float l0 = red[0][0] + drb[0], l1 = red[1][0] + drb[1], l2 = red[2][0] + drb[2];
    int chosen = 0; float best = l0;
    if (l1 > best) { best = l1; chosen = 1; }
    if (l2 > best) { best = l2; chosen = 2; }
    int steps = chosen < maxSteps ? chosen : maxSteps;
    ctlw[slot] = steps;
  }
}

// out = a + d*(b-a)   (torch.lerp)
__global__ void k_lerp(PRED, const float* __restrict__ a, const float* __restrict__ b,
                       float d, float* __restrict__ out, _Float16* outh) {
  PCHECK;
  int i = blockIdx.x * 256 + threadIdx.x;
  if (i >= HD) return;
  float v = a[i] + d * (b[i] - a[i]);
  out[i] = v;
  if (outh) outh[i] = (_Float16)v;
}

// cat1 = f16([ s0 - s1/2 (+ s2/3 if steps>=2) , s0 ])   (Ramanujan weights)
__global__ void k_cat1(PRED, const float* __restrict__ s0, const float* __restrict__ s1,
                       const float* __restrict__ s2, int stepsIdx,
                       _Float16* __restrict__ cat) {
  PCHECK;
  int i = blockIdx.x * 256 + threadIdx.x;
  if (i >= HD) return;
  int steps = ctl[stepsIdx];
  float agg = s0[i] - 0.5f * s1[i];
  if (steps >= 2) agg += s2[i] * (1.0f / 3.0f);
  cat[i] = (_Float16)agg;
  cat[HD + i] = (_Float16)s0[i];
}

__global__ void k_addb(PRED, const float* __restrict__ in, const float* __restrict__ b,
                       float* __restrict__ out) {
  PCHECK;
  int i = blockIdx.x * 256 + threadIdx.x;
  if (i < HD) out[i] = in[i] + b[i];
}

__global__ void k_cat2(PRED, const float* __restrict__ hcur, const float* __restrict__ anchor,
                       _Float16* __restrict__ cat) {
  PCHECK;
  int i = blockIdx.x * 256 + threadIdx.x;
  if (i >= HD) return;
  cat[i] = (_Float16)hcur[i];
  cat[HD + i] = (_Float16)anchor[i];
}

// g = sigmoid(agy + ag_b); out = g*anchor + (1-g)*hcur
__global__ void k_gate(PRED, const float* __restrict__ agy, const float* __restrict__ agb,
                       const float* __restrict__ anchor, const float* __restrict__ hcur,
                       float* __restrict__ out, _Float16* outh) {
  PCHECK;
  int i = blockIdx.x * 256 + threadIdx.x;
  if (i >= HD) return;
  float g = sigf(agy[i] + agb[i]);
  float v = g * anchor[i] + (1.0f - g) * hcur[i];
  out[i] = v;
  if (outh) outh[i] = (_Float16)v;
}

// ---------------------------------------------------------------------------
extern "C" void kernel_launch(void* const* d_in, const int* in_sizes, int n_in,
                              void* d_out, int out_size, void* d_ws, size_t ws_size,
                              hipStream_t stream) {
  const float* X    = (const float*)d_in[0];
  const float* Hin  = (const float*)d_in[1];
  const float* Cin  = (const float*)d_in[2];
  const float* Wv   = (const float*)d_in[5];
  const float* Wo   = (const float*)d_in[6];
  const float* Wih  = (const float*)d_in[7];
  const float* Whh  = (const float*)d_in[8];
  const float* BL   = (const float*)d_in[9];
  const float* RW   = (const float*)d_in[10];
  const float* RB   = (const float*)d_in[11];
  const float* EW   = (const float*)d_in[12];
  const float* EB   = (const float*)d_in[13];
  const float* DRW  = (const float*)d_in[14];
  const float* DRB  = (const float*)d_in[15];
  const float* RAMW = (const float*)d_in[16];
  const float* RAMB = (const float*)d_in[17];
  const float* AGW  = (const float*)d_in[18];
  const float* AGB  = (const float*)d_in[19];
  float* OUT = (float*)d_out;

  // ---- workspace layout (bump allocator, deterministic) ----
  char* base = (char*)d_ws;
  size_t off = 0;
  auto alloc = [&](size_t bytes) -> char* {
    char* p = base + off;
    off += (bytes + 255) & ~(size_t)255;
    return p;
  };
  int* ctl = (int*)alloc(64);
  // f16 swizzled weights (total ~144 MB, L2-resident working set)
  unsigned* WvH  = (unsigned*)alloc((size_t)HD * HD * 2);
  unsigned* WoH  = (unsigned*)alloc((size_t)HD * HD * 2);
  unsigned* WihH = (unsigned*)alloc((size_t)HD * HD4 * 2);
  unsigned* WhhH = (unsigned*)alloc((size_t)HD * HD4 * 2);
  unsigned* ExpH = (unsigned*)alloc((size_t)4 * HD * HD * 2);
  unsigned* RamH = (unsigned*)alloc((size_t)KK2 * HD * 2);
  unsigned* AgH  = (unsigned*)alloc((size_t)KK2 * HD * 2);

  _Float16* xh  = (_Float16*)alloc(HD * 2);
  _Float16* h0h = (_Float16*)alloc(HD * 2);

  struct Frame {
    float *attn1, *attn, *gates, *h2, *c2, *eo, *hp, *p;
    _Float16 *attn1h, *attnh, *h2h, *hph;
  };
  Frame F[5];
  for (int f = 0; f < 5; ++f) {
    F[f].attn1  = (float*)alloc(HD * 4);
    F[f].attn   = (float*)alloc(HD * 4);
    F[f].gates  = (float*)alloc(HD4 * 4);
    F[f].h2     = (float*)alloc(HD * 4);
    F[f].c2     = (float*)alloc(HD * 4);
    F[f].eo     = (float*)alloc(HD4 * 4);
    F[f].hp     = (float*)alloc(HD * 4);
    F[f].p      = (float*)alloc(4 * 4);
    F[f].attn1h = (_Float16*)alloc(HD * 2);
    F[f].attnh  = (_Float16*)alloc(HD * 2);
    F[f].h2h    = (_Float16*)alloc(HD * 2);
    F[f].hph    = (_Float16*)alloc(HD * 2);
  }
  // post-frame scratch
  float* rO1 = (float*)alloc(HD * 4);
  float* rH1 = (float*)alloc(HD * 4);
  float* rC1 = (float*)alloc(HD * 4);
  _Float16* rH1h = (_Float16*)alloc(HD * 2);
  float* rO3 = (float*)alloc(HD * 4);
  float* rH3 = (float*)alloc(HD * 4);
  float* rC3 = (float*)alloc(HD * 4);
  _Float16* rH3h = (_Float16*)alloc(HD * 2);
  float* hn1 = (float*)alloc(HD * 4);         // F1 lerp result
  _Float16* hn1h = (_Float16*)alloc(HD * 2);
  float* hn3 = (float*)alloc(HD * 4);         // F3 lerp result
  _Float16* hn3h = (_Float16*)alloc(HD * 2);
  float* h1 = (float*)alloc(HD * 4);          // F0 state after iter0
  _Float16* h1h = (_Float16*)alloc(HD * 2);
  float* h2n = (float*)alloc(HD * 4);         // F0 state after iter1
  float* f0_final = (float*)alloc(HD * 4);
  float* f0_ccur  = (float*)alloc(HD * 4);
  float* f0_hcur  = (float*)alloc(HD * 4);
  _Float16* cat1h0 = (_Float16*)alloc(2 * HD * 2);
  _Float16* cat1h1 = (_Float16*)alloc(2 * HD * 2);
  _Float16* cat1h3 = (_Float16*)alloc(2 * HD * 2);
  _Float16* cat2h0 = (_Float16*)alloc(2 * HD * 2);
  _Float16* cat2h1 = (_Float16*)alloc(2 * HD * 2);
  _Float16* cat2h3 = (_Float16*)alloc(2 * HD * 2);
  float* ramy = (float*)alloc(HD * 4);
  float* agy  = (float*)alloc(HD * 4);
  float* ramy0 = (float*)alloc(HD * 4);
  float* agy0  = (float*)alloc(HD * 4);
  float* anch0 = (float*)alloc(HD * 4);
  float* anch1 = (float*)alloc(HD * 4);
  float* anch3 = (float*)alloc(HD * 4);
  float* ramy3 = (float*)alloc(HD * 4);
  float* agy3  = (float*)alloc(HD * 4);

  if (ws_size < off) return;  // workspace too small: refuse (avoid OOB)

  // ---- launch helpers ----
  int GH  = (HD + 255) / 256;
  auto LZERO = [&](int ci, int lo, int hi, float* p, int n) {
    k_zero<<<(n + 255) / 256, 256, 0, stream>>>(ctl, ci, lo, hi, p, n);
  };
  auto LCVT = [&](int ci, int lo, int hi, const float* in, _Float16* out, int n) {
    k_cvt<<<(n + 255) / 256, 256, 0, stream>>>(ctl, ci, lo, hi, in, out, n);
  };
  auto LCOPY = [&](int ci, int lo, int hi, const float* in, float* out, _Float16* outh) {
    k_copy<<<GH, 256, 0, stream>>>(ctl, ci, lo, hi, in, out, outh, HD);
  };
  auto LGEMV = [&](int ci, int lo, int hi, const unsigned* Wt, const _Float16* v,
                   float* y, int K, int N, int kc) {
    int waves = (N >> 4) * kc;
    k_gemv<<<(waves * 32 + 255) / 256, 256, 0, stream>>>(ctl, ci, lo, hi, Wt, v, y, K, N, kc);
  };
  auto LCONV = [&](const float* W, unsigned* out, int K, int N) {
    int total = (K >> 1) * N;
    k_convert<<<(total + 255) / 256, 256, 0, stream>>>(W, out, K, N);
  };

  // ---- 0. control init + weight conversion (HBM sweep, once) ----
  k_ctlinit<<<1, 32, 0, stream>>>(ctl);
  LCONV(Wv, WvH, HD, HD);
  LCONV(Wo, WoH, HD, HD);
  LCONV(Wih, WihH, HD, HD4);
  LCONV(Whh, WhhH, HD, HD4);
  for (int e = 0; e < 4; ++e)
    LCONV(EW + (size_t)e * HD * HD, ExpH + (size_t)e * HD * HD / 2, HD, HD);
  LCONV(RAMW, RamH, KK2, HD);
  LCONV(AGW, AgH, KK2, HD);
  LCVT(-1, 0, 0, X, xh, HD);
  LCVT(-1, 0, 0, Hin, h0h, HD);

  // ---- pre(frame): cell_step + path_gate (+ depth router) ----
  auto PRE = [&](Frame& Fr, int ci, int lo, int hi, const _Float16* hbf,
                 const float* cin, int drSlot, int maxSteps) {
    LZERO(ci, lo, hi, Fr.attn1, HD);
    LGEMV(ci, lo, hi, WvH, hbf, Fr.attn1, HD, HD, 8);
    LCVT(ci, lo, hi, Fr.attn1, Fr.attn1h, HD);
    LZERO(ci, lo, hi, Fr.attn, HD);
    LGEMV(ci, lo, hi, WoH, Fr.attn1h, Fr.attn, HD, HD, 8);
    LCVT(ci, lo, hi, Fr.attn, Fr.attnh, HD);
    LZERO(ci, lo, hi, Fr.gates, HD4);
    LGEMV(ci, lo, hi, WihH, xh, Fr.gates, HD, HD4, 4);
    LGEMV(ci, lo, hi, WhhH, Fr.attnh, Fr.gates, HD, HD4, 4);
    k_lstm<<<GH, 256, 0, stream>>>(ctl, ci, lo, hi, Fr.gates, BL, cin, Fr.h2, Fr.h2h, Fr.c2);
    k_router<<<1, 256, 0, stream>>>(ctl, ci, lo, hi, Fr.h2, RW, RB, Fr.p);
    LZERO(ci, lo, hi, Fr.eo, HD4);
    for (int e = 0; e < 4; ++e)
      LGEMV(ci, lo, hi, ExpH + (size_t)e * HD * HD / 2, Fr.h2h, Fr.eo + e * HD, HD, HD, 8);
    k_expcomb<<<GH, 256, 0, stream>>>(ctl, ci, lo, hi, Fr.eo, Fr.p, EB, Fr.hp, Fr.hph);
    if (drSlot >= 0)
      k_depth<<<1, 256, 0, stream>>>(ctl, ci, lo, hi, Fr.hp, DRW, DRB, maxSteps, drSlot, ctl);
  };

  // ---- post(mid frame): lerp + ramanujan + anchor gate + output select ----
  auto POSTMID = [&](int slot, Frame& Fr, const float* chRO, const float* chRH,
                     const float* chRC, float damping,
                     float* hnext, _Float16* hnexth, _Float16* cat1h, float* ry,
                     float* anch, _Float16* cat2h, float* ay,
                     float* rO, float* rH, _Float16* rHh, float* rC) {
    // active path (steps >= 1)
    k_lerp<<<GH, 256, 0, stream>>>(ctl, slot, 1, BIG, Fr.hp, chRH, damping, hnext, hnexth);
    k_cat1<<<GH, 256, 0, stream>>>(ctl, slot, 1, BIG, Fr.hp, hnext, hnext, slot, cat1h);
    LZERO(slot, 1, BIG, ry, HD);
    LGEMV(slot, 1, BIG, RamH, cat1h, ry, KK2, HD, 8);
    k_addb<<<GH, 256, 0, stream>>>(ctl, slot, 1, BIG, ry, RAMB, anch);
    k_cat2<<<GH, 256, 0, stream>>>(ctl, slot, 1, BIG, hnext, anch, cat2h);
    LZERO(slot, 1, BIG, ay, HD);
    LGEMV(slot, 1, BIG, AgH, cat2h, ay, KK2, HD, 8);
    k_gate<<<GH, 256, 0, stream>>>(ctl, slot, 1, BIG, ay, AGB, anch, hnext, rH, rHh);
    LCOPY(slot, 1, BIG, chRO, rO, nullptr);
    LCOPY(slot, 1, BIG, chRC, rC, nullptr);
    // inactive path (steps == 0): return (so, hp, cs)
    LCOPY(slot, 0, 0, Fr.h2, rO, nullptr);
    LCOPY(slot, 0, 0, Fr.hp, rH, rHh);
    LCOPY(slot, 0, 0, Fr.c2, rC, nullptr);
  };

  // ================== unrolled worst-case call tree ==================
  // ctl[0]=steps(F0,depth0)  ctl[1]=steps(F1,depth1)  ctl[2]=steps(F3,depth1)

  // F0 = fwd(depth 0), always active
  PRE(F[0], -1, 0, 0, h0h, Cin, /*drSlot=*/0, /*maxSteps=*/2);

  // iter0 of F0: F1 = fwd(depth 1) on (x, hp0, c2_0), active if ctl0>=1
  PRE(F[1], 0, 1, BIG, F[0].hph, F[0].c2, /*drSlot=*/1, /*maxSteps=*/1);
  // F2 = fwd(depth 2) leaf on (x, hp1, c2_1), active if ctl1>=1
  PRE(F[2], 1, 1, BIG, F[1].hph, F[1].c2, -1, 0);
  // F1 post: damping at depth 1 = 1/(1+1)^2 = 0.25; leaf returns (h2_2, hp2, c2_2)
  POSTMID(1, F[1], F[2].h2, F[2].hp, F[2].c2, 0.25f,
          hn1, hn1h, cat1h1, ramy, anch1, cat2h1, agy, rO1, rH1, rH1h, rC1);

  // F0 state after iter0: damping at depth0 = 1.0 -> h1 = rH1
  k_lerp<<<GH, 256, 0, stream>>>(ctl, 0, 1, BIG, F[0].hp, rH1, 1.0f, h1, h1h);

  // iter1 of F0: F3 = fwd(depth 1) on (x, h1, rC1), active if ctl0>=2
  PRE(F[3], 0, 2, BIG, h1h, rC1, /*drSlot=*/2, /*maxSteps=*/1);
  // F4 = fwd(depth 2) leaf, active if ctl2>=1
  PRE(F[4], 2, 1, BIG, F[3].hph, F[3].c2, -1, 0);
  POSTMID(2, F[3], F[4].h2, F[4].hp, F[4].c2, 0.25f,
          hn3, hn3h, cat1h3, ramy3, anch3, cat2h3, agy3, rO3, rH3, rH3h, rC3);

  // F0 state after iter1
  k_lerp<<<GH, 256, 0, stream>>>(ctl, 0, 2, BIG, h1, rH3, 1.0f, h2n, nullptr);

  // F0 final / c_cur / h_cur selection by steps0
  LCOPY(0, 1, 1, rO1, f0_final, nullptr);
  LCOPY(0, 1, 1, rC1, f0_ccur, nullptr);
  LCOPY(0, 1, 1, h1, f0_hcur, nullptr);
  LCOPY(0, 2, BIG, rO3, f0_final, nullptr);
  LCOPY(0, 2, BIG, rC3, f0_ccur, nullptr);
  LCOPY(0, 2, BIG, h2n, f0_hcur, nullptr);

  // F0 post (ramanujan over [hp0, h1, (h2n)] + anchor gate), active if ctl0>=1
  k_cat1<<<GH, 256, 0, stream>>>(ctl, 0, 1, BIG, F[0].hp, h1, h2n, 0, cat1h0);
  LZERO(0, 1, BIG, ramy0, HD);
  LGEMV(0, 1, BIG, RamH, cat1h0, ramy0, KK2, HD, 8);
  k_addb<<<GH, 256, 0, stream>>>(ctl, 0, 1, BIG, ramy0, RAMB, anch0);
  k_cat2<<<GH, 256, 0, stream>>>(ctl, 0, 1, BIG, f0_hcur, anch0, cat2h0);
  LZERO(0, 1, BIG, agy0, HD);
  LGEMV(0, 1, BIG, AgH, cat2h0, agy0, KK2, HD, 8);
  k_gate<<<GH, 256, 0, stream>>>(ctl, 0, 1, BIG, agy0, AGB, anch0, f0_hcur,
                                 OUT + HD, nullptr);
  LCOPY(0, 1, BIG, f0_final, OUT, nullptr);
  LCOPY(0, 1, BIG, f0_ccur, OUT + 2 * HD, nullptr);

  // steps0 == 0: return (so0, hp0, cs0) directly
  LCOPY(0, 0, 0, F[0].h2, OUT, nullptr);
  LCOPY(0, 0, 0, F[0].hp, OUT + HD, nullptr);
  LCOPY(0, 0, 0, F[0].c2, OUT + 2 * HD, nullptr);
}
